// randlanet_42597485642042
// MI455X (gfx1250) — compile-verified, matrix operand order checked by reading, not run on hardware
//
#include <hip/hip_runtime.h>

typedef __attribute__((ext_vector_type(16))) _Float16 v16h;
typedef __attribute__((ext_vector_type(8)))  float    v8f;

#define BN_EPS 1e-6f

__device__ __forceinline__ v8f wmma_f16(const v16h& a, const v16h& b, const v8f& c) {
  return __builtin_amdgcn_wmma_f32_16x16x32_f16(false, a, false, b, (short)0, c, false, false);
}

// Fill B fragment: column colC (clamped, always valid), rows k = kc + pattern.
__device__ __forceinline__ v16h load_b_frag(const float* __restrict__ Wc, long long dout,
                                            int kc, int koff) {
  v16h b;
#pragma unroll
  for (int h = 0; h < 16; ++h) {
    const int k = kc + ((h < 8) ? (koff + h) : (16 + koff + (h - 8)));
    b[h] = (_Float16)Wc[(long long)k * dout];
  }
  return b;
}

// ---------------------------------------------------------------------------
// Generic per-point linear (1x1 conv) + optional BN + optional leakyReLU.
// X:[P,din] row-major, W:[din,dout], Y:[P,dout].
// One wave per 16(points)x16(channels) tile, K-chunks of 32 via f16 WMMA.
// flags bit0 = BN, bit1 = leakyReLU(0.2)
// ---------------------------------------------------------------------------
__global__ __launch_bounds__(256) void k_gemm_bn(
    const float* __restrict__ X, const float* __restrict__ W,
    const float* __restrict__ bias,
    const float* __restrict__ gg, const float* __restrict__ bb,
    const float* __restrict__ mm, const float* __restrict__ vv,
    float* __restrict__ Y, int P, int din, int dout, int flags)
{
  const int tilesN = (dout + 15) >> 4;
  const int tilesM = (P + 15) >> 4;
  const long long totalTiles = (long long)tilesM * tilesN;
  const int lane = threadIdx.x & 31;
  const long long gw = (long long)blockIdx.x * 8 + (threadIdx.x >> 5);
  if (gw >= totalTiles) return;                   // wave-uniform: EXEC stays all-1s
  const int tm = (int)(gw / tilesN);
  const int tn = (int)(gw % tilesN);
  const int row  = tm * 16 + (lane & 15);
  const int col  = tn * 16 + (lane & 15);
  const bool colOK = col < dout;
  const int rowC = (row < P) ? row : (P - 1);     // clamp: loads always in-bounds
  const int colC = colOK ? col : (dout - 1);
  const int koff = (lane & 16) ? 8 : 0;

  const float* __restrict__ Xr = X + (long long)rowC * din;
  const float* __restrict__ Wc = W + colC;

  v8f acc = {0.f, 0.f, 0.f, 0.f, 0.f, 0.f, 0.f, 0.f};
  int kc = 0;
  if ((din & 3) == 0) {
    // vectorized A path: two contiguous 8-float runs per lane -> 4x b128 loads
    for (; kc + 32 <= din; kc += 32) {
      const float4 a0 = *(const float4*)(Xr + kc + koff);
      const float4 a1 = *(const float4*)(Xr + kc + koff + 4);
      const float4 a2 = *(const float4*)(Xr + kc + 16 + koff);
      const float4 a3 = *(const float4*)(Xr + kc + 16 + koff + 4);
      v16h a;
      a[0]  = (_Float16)a0.x; a[1]  = (_Float16)a0.y; a[2]  = (_Float16)a0.z; a[3]  = (_Float16)a0.w;
      a[4]  = (_Float16)a1.x; a[5]  = (_Float16)a1.y; a[6]  = (_Float16)a1.z; a[7]  = (_Float16)a1.w;
      a[8]  = (_Float16)a2.x; a[9]  = (_Float16)a2.y; a[10] = (_Float16)a2.z; a[11] = (_Float16)a2.w;
      a[12] = (_Float16)a3.x; a[13] = (_Float16)a3.y; a[14] = (_Float16)a3.z; a[15] = (_Float16)a3.w;
      const v16h b = load_b_frag(Wc, dout, kc, koff);
      acc = wmma_f16(a, b, acc);
    }
  } else {
    for (; kc + 32 <= din; kc += 32) {
      v16h a;
#pragma unroll
      for (int h = 0; h < 16; ++h) {
        const int k = kc + ((h < 8) ? (koff + h) : (16 + koff + (h - 8)));
        a[h] = (_Float16)Xr[k];
      }
      const v16h b = load_b_frag(Wc, dout, kc, koff);
      acc = wmma_f16(a, b, acc);
    }
  }
  if (kc < din) {
    // tail chunk: unconditional clamped loads + cndmask zeroing (no exec branches)
    v16h a, b;
#pragma unroll
    for (int h = 0; h < 16; ++h) {
      const int k = kc + ((h < 8) ? (koff + h) : (16 + koff + (h - 8)));
      const int kcl = (k < din) ? k : (din - 1);
      float av = Xr[kcl];
      float bv = Wc[(long long)kcl * dout];
      av = (k < din) ? av : 0.0f;
      bv = (k < din) ? bv : 0.0f;
      a[h] = (_Float16)av;
      b[h] = (_Float16)bv;
    }
    acc = wmma_f16(a, b, acc);
  }

  float scale = 1.0f, shift = 0.0f;
  if (colOK) {
    shift = bias ? bias[col] : 0.0f;
    if (flags & 1) {
      const float s = gg[col] * rsqrtf(vv[col] + BN_EPS);
      shift = (shift - mm[col]) * s + bb[col];
      scale = s;
    }
  }
  const int rbase = tm * 16 + ((lane & 16) ? 8 : 0);
#pragma unroll
  for (int r = 0; r < 8; ++r) {
    const int m = rbase + r;
    if (m < P && colOK) {
      float y = acc[r] * scale + shift;
      if (flags & 2) y = (y > 0.0f) ? y : 0.2f * y;
      Y[(long long)m * dout + col] = y;
    }
  }
}

// ---------------------------------------------------------------------------
// Relative position encoding: [dist, rel(3), center(3), neighbor(3)] per (p,k)
// ---------------------------------------------------------------------------
__global__ void k_rpe(const float* __restrict__ xyz, const int* __restrict__ neigh,
                      float* __restrict__ rpe, int Npts, int Nper)
{
  const long long t = (long long)blockIdx.x * blockDim.x + threadIdx.x;
  const long long total = (long long)Npts * 16;
  if (t >= total) return;
  const long long p = t >> 4;
  const int b = (int)(p / Nper);
  const int j = neigh[t];
  const float* c  = xyz + p * 3;
  const float* nb = xyz + ((long long)b * Nper + j) * 3;
  const float rx = c[0] - nb[0], ry = c[1] - nb[1], rz = c[2] - nb[2];
  const float dist = sqrtf(rx * rx + ry * ry + rz * rz + 1e-12f);
  float* o = rpe + t * 10;
  o[0] = dist; o[1] = rx;  o[2] = ry;  o[3] = rz;
  o[4] = c[0]; o[5] = c[1]; o[6] = c[2];
  o[7] = nb[0]; o[8] = nb[1]; o[9] = nb[2];
}

// ---------------------------------------------------------------------------
// Fused attention pool: one workgroup per point.
// cat[k][i] = (i<dh) ? featSrc[b, neigh[p,k], i] : fx[p,k,i-dh]   (d = 2*dh)
// s = cat(16xd) @ attw(dxd) via WMMA; softmax over k; pooled[e]=sum_k cat*p
// ---------------------------------------------------------------------------
__global__ __launch_bounds__(256) void k_att_pool(
    const float* __restrict__ featSrc, const int* __restrict__ neigh,
    const float* __restrict__ fx, const float* __restrict__ attw,
    float* __restrict__ pooled, int Npts, int Nper, int dh)
{
  __shared__ float catb[16 * 256];
  __shared__ float sbuf[16 * 256];
  const int p = blockIdx.x;
  if (p >= Npts) return;
  const int d = dh * 2;
  const int tid = threadIdx.x;
  const int b = p / Nper;

  for (int idx = tid; idx < 16 * d; idx += 256) {
    const int k = idx / d, i = idx - k * d;
    float v;
    if (i < dh) {
      const int j = neigh[(long long)p * 16 + k];
      v = featSrc[((long long)b * Nper + j) * dh + i];
    } else {
      v = fx[((long long)p * 16 + k) * dh + (i - dh)];
    }
    catb[k * d + i] = v;
  }
  __syncthreads();

  const int wave = tid >> 5, lane = tid & 31;
  const int tilesN = d >> 4;                       // d in {16,64,128,256}
  const int rowk = lane & 15;
  const int koff = (lane & 16) ? 8 : 0;
  for (int tn = wave; tn < tilesN; tn += 8) {      // wave-uniform loop
    const int col = tn * 16 + (lane & 15);
    v8f acc = {0.f, 0.f, 0.f, 0.f, 0.f, 0.f, 0.f, 0.f};
    int kc = 0;
    for (; kc + 32 <= d; kc += 32) {               // full chunks (d=64/128/256)
      v16h a, bf;
#pragma unroll
      for (int h = 0; h < 16; ++h) {
        const int kk = kc + ((h < 8) ? (koff + h) : (16 + koff + (h - 8)));
        a[h]  = (_Float16)catb[rowk * d + kk];
        bf[h] = (_Float16)attw[(long long)kk * d + col];
      }
      acc = wmma_f16(a, bf, acc);
    }
    if (kc < d) {                                  // tail (only d==16)
      v16h a, bf;
#pragma unroll
      for (int h = 0; h < 16; ++h) {
        const int kk = kc + ((h < 8) ? (koff + h) : (16 + koff + (h - 8)));
        const int kcl = (kk < d) ? kk : (d - 1);
        float av = catb[rowk * d + kcl];
        float bv = attw[(long long)kcl * d + col];
        av = (kk < d) ? av : 0.0f;
        bv = (kk < d) ? bv : 0.0f;
        a[h]  = (_Float16)av;
        bf[h] = (_Float16)bv;
      }
      acc = wmma_f16(a, bf, acc);
    }
    const int rb = (lane & 16) ? 8 : 0;
#pragma unroll
    for (int r = 0; r < 8; ++r) sbuf[(rb + r) * d + col] = acc[r];
  }
  __syncthreads();

  for (int e = tid; e < d; e += 256) {
    float mx = sbuf[e];
    for (int k = 1; k < 16; ++k) mx = fmaxf(mx, sbuf[k * d + e]);
    float sum = 0.0f, ws = 0.0f;
    for (int k = 0; k < 16; ++k) {
      const float pe = expf(sbuf[k * d + e] - mx);
      sum += pe;
      ws  += pe * catb[k * d + e];
    }
    pooled[(long long)p * d + e] = ws / sum;
  }
}

// ---------------------------------------------------------------------------
// random_sample: out[b,m,e] = max_k fe[b, sub[b,m,k], e]
// ---------------------------------------------------------------------------
__global__ void k_gather_max(const float* __restrict__ fe, const int* __restrict__ sub,
                             float* __restrict__ out, int Mpts, int Mper, int Nper, int d)
{
  const long long t = (long long)blockIdx.x * blockDim.x + threadIdx.x;
  const long long total = (long long)Mpts * d;
  if (t >= total) return;
  const int e = (int)(t % d);
  const long long pm = t / d;
  const int b = (int)(pm / Mper);
  const int* si = sub + pm * 16;
  float mx = -3.4e38f;
  for (int k = 0; k < 16; ++k) {
    const int j = si[k];
    mx = fmaxf(mx, fe[((long long)b * Nper + j) * d + e]);
  }
  out[t] = mx;
}

// ---------------------------------------------------------------------------
// Decoder concat: out[p, 0:d1] = enc[p,:]; out[p, d1:] = f[b, interp[p], :]
// ---------------------------------------------------------------------------
__global__ void k_interp_concat(const float* __restrict__ enc, const float* __restrict__ f,
                                const int* __restrict__ interp, float* __restrict__ out,
                                int Npts, int Nper, int Mper, int d1, int d2)
{
  const int dt = d1 + d2;
  const long long t = (long long)blockIdx.x * blockDim.x + threadIdx.x;
  const long long total = (long long)Npts * dt;
  if (t >= total) return;
  const int c = (int)(t % dt);
  const long long p = t / dt;
  float v;
  if (c < d1) {
    v = enc[p * d1 + c];
  } else {
    const int b = (int)(p / Nper);
    const int j = interp[p];
    v = f[((long long)b * Mper + j) * d2 + (c - d1)];
  }
  out[t] = v;
}

__global__ void k_add_lrelu(const float* __restrict__ a, const float* __restrict__ b,
                            float* __restrict__ y, long long n)
{
  const long long t = (long long)blockIdx.x * blockDim.x + threadIdx.x;
  if (t >= n) return;
  const float v = a[t] + b[t];
  y[t] = (v > 0.0f) ? v : 0.2f * v;
}

// in:[B*N, C] -> out:[B, C, N]
__global__ void k_transpose_out(const float* __restrict__ in, float* __restrict__ out,
                                int Bn, int Nper, int C)
{
  const long long t = (long long)blockIdx.x * blockDim.x + threadIdx.x;
  const long long total = (long long)Bn * Nper * C;
  if (t >= total) return;
  const int c = (int)(t % C);
  const long long pn = t / C;
  const int b = (int)(pn / Nper);
  const int n = (int)(pn % Nper);
  out[((long long)b * C + c) * Nper + n] = in[t];
}

// ---------------------------------------------------------------------------
// Host side
// ---------------------------------------------------------------------------
struct ConvP { const float *b, *be, *g, *m, *v, *w; };
struct AttP  { const float* w; ConvP mlp; };
struct BlockP { AttP att1, att2; ConvP mlp1, mlp2, shortc, xyz1, xyz2; };

extern "C" void kernel_launch(void* const* d_in, const int* in_sizes, int n_in,
                              void* d_out, int out_size, void* d_ws, size_t ws_size,
                              hipStream_t stream) {
  (void)in_sizes; (void)n_in; (void)out_size; (void)ws_size;

  static const int Bsz = 4;
  static const int NS[5]   = {45056, 11264, 2816, 704, 176};
  static const int DOUT[4] = {16, 64, 128, 256};

  // ---- inputs (setup_inputs dict order) ----
  const float* xyzA[4]; const int* neighA[4]; const int* subA[4]; const int* interpA[4];
  for (int i = 0; i < 4; ++i) {
    xyzA[i]    = (const float*)d_in[4 * i + 0];
    neighA[i]  = (const int*)  d_in[4 * i + 1];
    subA[i]    = (const int*)  d_in[4 * i + 2];
    interpA[i] = (const int*)  d_in[4 * i + 3];
  }
  const float* features = (const float*)d_in[16];

  // ---- params: jax pytree (dict keys sorted alphabetically) ----
  int pi = 17;
  auto nxt = [&]() { return (const float*)d_in[pi++]; };
  auto getConv = [&]() { ConvP c; c.b = nxt(); c.be = nxt(); c.g = nxt(); c.m = nxt(); c.v = nxt(); c.w = nxt(); return c; };
  auto getAtt  = [&]() { AttP a; a.w = nxt(); a.mlp = getConv(); return a; };

  BlockP blk[4];
  for (int i = 0; i < 4; ++i) {           // block keys: att1,att2,mlp1,mlp2,short,xyz1,xyz2
    blk[i].att1 = getAtt();  blk[i].att2 = getAtt();
    blk[i].mlp1 = getConv(); blk[i].mlp2 = getConv();
    blk[i].shortc = getConv();
    blk[i].xyz1 = getConv(); blk[i].xyz2 = getConv();
  }
  const float* bn0_be = nxt(); const float* bn0_g = nxt();
  const float* bn0_m  = nxt(); const float* bn0_v = nxt();
  ConvP dec[4]; for (int j = 0; j < 4; ++j) dec[j] = getConv();
  ConvP dec0 = getConv();
  const float* fc_b  = nxt(); const float* fc_w  = nxt();
  const float* fc0_b = nxt(); const float* fc0_w = nxt();
  ConvP fc1 = getConv();
  ConvP fc2 = getConv();

  // ---- workspace carve (floats) ----
  float* wsf = (float*)d_ws;
  size_t off = 0;
  auto alloc = [&](size_t n) { float* p = wsf + off; off += n; return p; };
  float* enc0 = alloc(5767168);   // [B,N0,32]
  float* enc1 = alloc(1441792);   // [B,N1,32]
  float* enc2 = alloc(1441792);   // [B,N2,128]
  float* enc3 = alloc(720896);    // [B,N3,256]
  float* enc4 = alloc(360448);    // [B,N4,512]
  float* fbufA = alloc(11534336);
  float* fbufB = alloc(11534336);
  float* nb1 = alloc(28835840);   // rpe / fx2
  float* nb2 = alloc(23068672);   // fx
  float* pcbuf   = alloc(1441792);
  float* aggbuf  = alloc(1441792);
  float* poolbuf = alloc(2883584);
  float* agg2buf = alloc(2883584);
  float* m2buf = alloc(5767168);
  float* shbuf = alloc(5767168);
  float* encArr[5] = {enc0, enc1, enc2, enc3, enc4};

  auto gemm = [&](const float* X, const float* W, const float* bias,
                  const float* g, const float* be, const float* m, const float* v,
                  float* Y, long long P, int din, int dout, int flags) {
    const long long waves = ((P + 15) / 16) * (long long)((dout + 15) / 16);
    const int blocks = (int)((waves + 7) / 8);
    k_gemm_bn<<<blocks, 256, 0, stream>>>(X, W, bias, g, be, m, v, Y, (int)P, din, dout, flags);
  };
  auto gemmC = [&](const float* X, const ConvP& c, float* Y, long long P, int din, int dout, int flags) {
    gemm(X, c.w, c.b, c.g, c.be, c.m, c.v, Y, P, din, dout, flags);
  };
  auto gridFor = [](long long n) { return dim3((unsigned)((n + 255) / 256)); };

  // ================= encoder =================
  // fc0 + bn0 + lrelu
  gemm(features, fc0_w, fc0_b, bn0_g, bn0_be, bn0_m, bn0_v, fbufA,
       (long long)Bsz * NS[0], 3, 8, 3);

  const float* feat = fbufA;
  int dinCur = 8;
  for (int i = 0; i < 4; ++i) {
    const int N = NS[i];
    const long long P = (long long)Bsz * N;
    const long long PK = P * 16;
    const int dout = DOUT[i], dh = dout / 2, din = dinCur;
    const BlockP& bp = blk[i];

    gemmC(feat, bp.mlp1, pcbuf, P, din, dh, 3);                              // pc
    k_rpe<<<gridFor(PK), 256, 0, stream>>>(xyzA[i], neighA[i], nb1, (int)P, N);
    gemmC(nb1, bp.xyz1, nb2, PK, 10, dh, 3);                                 // fx
    k_att_pool<<<(unsigned)P, 256, 0, stream>>>(pcbuf, neighA[i], nb2, bp.att1.w,
                                                poolbuf, (int)P, N, dh);     // att1
    gemmC(poolbuf, bp.att1.mlp, aggbuf, P, dout, dh, 3);                     // agg
    gemmC(nb2, bp.xyz2, nb1, PK, dh, dh, 3);                                 // fx2
    k_att_pool<<<(unsigned)P, 256, 0, stream>>>(aggbuf, neighA[i], nb1, bp.att2.w,
                                                poolbuf, (int)P, N, dh);     // att2
    gemmC(poolbuf, bp.att2.mlp, agg2buf, P, dout, dout, 3);                  // agg2
    gemmC(agg2buf, bp.mlp2, m2buf, P, dout, 2 * dout, 1);                    // mlp2 (no act)
    gemmC(feat, bp.shortc, shbuf, P, din, 2 * dout, 1);                      // short (no act)

    float* fe = (i == 0) ? enc0 : fbufB;
    k_add_lrelu<<<gridFor(P * 2 * dout), 256, 0, stream>>>(m2buf, shbuf, fe, P * 2 * dout);

    const long long Mtot = (long long)Bsz * NS[i + 1] * 2 * dout;
    k_gather_max<<<gridFor(Mtot), 256, 0, stream>>>(fe, subA[i], encArr[i + 1],
                                                    Bsz * NS[i + 1], NS[i + 1], N, 2 * dout);
    feat = encArr[i + 1];
    dinCur = 2 * dout;
  }

  // ================= decoder =================
  gemmC(enc4, dec0, fbufA, (long long)Bsz * NS[4], 512, 512, 3);

  const int   encD[4]   = {256, 128, 32, 32};
  const float* encPtr[4] = {enc3, enc2, enc1, enc0};
  const int   decOut[4] = {256, 128, 32, 32};
  int curD = 512;
  for (int j = 0; j < 4; ++j) {
    const int lvlN = NS[3 - j], lvlM = NS[4 - j];
    const long long P = (long long)Bsz * lvlN;
    const int d1 = encD[j], d2 = curD;
    k_interp_concat<<<gridFor(P * (d1 + d2)), 256, 0, stream>>>(
        encPtr[j], fbufA, interpA[3 - j], fbufB, (int)P, lvlN, lvlM, d1, d2);
    gemmC(fbufB, dec[j], fbufA, P, d1 + d2, decOut[j], 3);
    curD = decOut[j];
  }

  const long long P0 = (long long)Bsz * NS[0];
  gemmC(fbufA, fc1, fbufB, P0, 32, 64, 3);
  gemmC(fbufB, fc2, fbufA, P0, 64, 32, 3);
  gemm(fbufA, fc_w, fc_b, nullptr, nullptr, nullptr, nullptr, fbufB, P0, 32, 19, 0);
  k_transpose_out<<<gridFor(P0 * 19), 256, 0, stream>>>(fbufB, (float*)d_out, Bsz, NS[0], 19);
}
